// MultiHeadAttentionClassifier_46377056862583
// MI455X (gfx1250) — matrix operand order repaired and checked by reading, not compiled, not run on hardware
//
#include <hip/hip_runtime.h>
#include <math.h>

#define Bv 8
#define Sv 2048
#define Dv 512
#define Hv 8
#define HDv 64
#define BSv (Bv*Sv)   /* 16384 tokens */
#define C2 2
#define QBLK 64       /* queries per attention workgroup (4 waves x 16) */

typedef __attribute__((ext_vector_type(16))) __bf16 v16bf;
typedef __attribute__((ext_vector_type(8)))  float  v8f;
typedef __attribute__((ext_vector_type(4)))  unsigned int u32x4;
typedef __attribute__((ext_vector_type(4)))  int v4i;

#if __has_builtin(__builtin_amdgcn_global_load_async_to_lds_b128) && \
    __has_builtin(__builtin_amdgcn_s_wait_asynccnt)
#define HAVE_ASYNC 1
#else
#define HAVE_ASYNC 0
#endif

// Builtin expects int4* in AS(1) (printed '__device__') / AS(3) ('__shared__').
#define GPTR(p) ((__attribute__((address_space(1))) v4i*)(void*)(p))
#define LPTR(p) ((__attribute__((address_space(3))) v4i*)(void*)(p))

union FragU { u32x4 u[2]; v16bf v; };

// Per-lane fragment load: two 16-byte chunks of 8 bf16 each.
// A (16x32) per lane l: m=l&15, chunks at k+8*(l>>4) and k+16+8*(l>>4).
// B (32x16) per lane l: n=l&15, chunks at k+8*(l>>4) and k+16+8*(l>>4).
__device__ __forceinline__ v16bf load_frag(const __bf16* row, int c0, int c1) {
  FragU f;
  f.u[0] = *(const u32x4*)(row + c0);
  f.u[1] = *(const u32x4*)(row + c1);
  return f.v;
}

__global__ void zero_kernel(float* p, int n) {
  int i = blockIdx.x * 256 + threadIdx.x;
  if (i < n) p[i] = 0.f;
}

__global__ void cvt_kernel(const float* __restrict__ src, __bf16* __restrict__ dst, int n) {
  int i = blockIdx.x * 256 + threadIdx.x;
  if (i < n) dst[i] = (__bf16)src[i];
}

// h = (emb[x] + pe) * sqrt(D); pad-bias = (sum_d h == 0) ? -inf : 0
__global__ void embed_kernel(const int* __restrict__ x, const float* __restrict__ emb,
                             const float* __restrict__ pe, __bf16* __restrict__ hbf,
                             float* __restrict__ padbias) {
  const int tok = blockIdx.x;                 // [0, B*S)
  const int s = tok & (Sv - 1);
  const int tid = threadIdx.x;                // 128 threads
  const int v = x[tok];
  const float sc = 22.62741699796952f;        // sqrt(512)
  float sum = 0.f;
  for (int d = tid; d < Dv; d += 128) {
    float val = (emb[(long)v * Dv + d] + pe[s * Dv + d]) * sc;
    hbf[(long)tok * Dv + d] = (__bf16)val;
    sum += val;
  }
  __shared__ float red[128];
  red[tid] = sum;
  __syncthreads();
  for (int st = 64; st; st >>= 1) {
    if (tid < st) red[tid] += red[tid + st];
    __syncthreads();
  }
  if (tid == 0) padbias[tok] = (red[0] == 0.0f) ? -__builtin_inff() : 0.0f;
}

// Q/K/V = h @ W^T + b. One wave per 16x64 output tile (A-frag reused over 4 B
// tiles -> 10 b128 loads per 4 wmma). Q,K row-major [B*S,D]; V transposed [B,D,S].
__global__ void __launch_bounds__(128) qkv_kernel(const __bf16* __restrict__ hbf,
    const __bf16* __restrict__ wq, const __bf16* __restrict__ wk, const __bf16* __restrict__ wv,
    const float* __restrict__ bq, const float* __restrict__ bk, const float* __restrict__ bv,
    __bf16* __restrict__ Qo, __bf16* __restrict__ Ko, __bf16* __restrict__ VTo)
{
  const int lane = threadIdx.x & 31;
  const int wave = threadIdx.x >> 5;
  const int g = lane >> 4, ln = lane & 15;
  const int tile = blockIdx.x * 4 + wave;
  const int tilesPerMat = (BSv / 16) * (Dv / 64);   // 8192
  const int which = tile / tilesPerMat;              // 0=Q 1=K 2=V (wave-uniform)
  const int t2 = tile - which * tilesPerMat;
  const int mi = t2 >> 3;
  const int ni = t2 & 7;
  const int m0 = mi * 16, n0 = ni * 64;

  const __bf16* W  = (which == 0) ? wq : (which == 1) ? wk : wv;
  const float*  bb = (which == 0) ? bq : (which == 1) ? bk : bv;
  const __bf16* arow = hbf + (m0 + ln) * Dv;
  const __bf16* brow = W   + (n0 + ln) * Dv;

  v8f a0 = {}, a1 = {}, a2 = {}, a3 = {};
  for (int kk = 0; kk < Dv; kk += 32) {
    v16bf a = load_frag(arow, kk + g * 8, kk + 16 + g * 8);
    v16bf b0 = load_frag(brow,            kk + g * 8, kk + 16 + g * 8);
    v16bf b1 = load_frag(brow + 16 * Dv,  kk + g * 8, kk + 16 + g * 8);
    v16bf b2 = load_frag(brow + 32 * Dv,  kk + g * 8, kk + 16 + g * 8);
    v16bf b3 = load_frag(brow + 48 * Dv,  kk + g * 8, kk + 16 + g * 8);
    a0 = __builtin_amdgcn_wmma_f32_16x16x32_bf16(false, a, false, b0, (short)0, a0, false, false);
    a1 = __builtin_amdgcn_wmma_f32_16x16x32_bf16(false, a, false, b1, (short)0, a1, false, false);
    a2 = __builtin_amdgcn_wmma_f32_16x16x32_bf16(false, a, false, b2, (short)0, a2, false, false);
    a3 = __builtin_amdgcn_wmma_f32_16x16x32_bf16(false, a, false, b3, (short)0, a3, false, false);
  }
  if (which < 2) {
    __bf16* O = (which == 0) ? Qo : Ko;
    #pragma unroll
    for (int j = 0; j < 4; ++j) {
      const v8f& ac = (j == 0) ? a0 : (j == 1) ? a1 : (j == 2) ? a2 : a3;
      float bias = bb[n0 + j * 16 + ln];
      #pragma unroll
      for (int r = 0; r < 8; ++r) {
        int m = m0 + r + 8 * g;                 // C/D layout: M = r + 8*(lane>=16)
        O[m * Dv + n0 + j * 16 + ln] = (__bf16)(ac[r] + bias);
      }
    }
  } else {
    int m = m0 + 8 * g;                          // 8 consecutive tokens, same batch
    int bidx = m >> 11;                          // m / S
    int s = m - (bidx << 11);
    #pragma unroll
    for (int j = 0; j < 4; ++j) {
      const v8f& ac = (j == 0) ? a0 : (j == 1) ? a1 : (j == 2) ? a2 : a3;
      float bias = bb[n0 + j * 16 + ln];
      union { u32x4 u; __bf16 e[8]; } pk;
      #pragma unroll
      for (int r = 0; r < 8; ++r) pk.e[r] = (__bf16)(ac[r] + bias);
      *(u32x4*)(VTo + (bidx * Dv + n0 + j * 16 + ln) * Sv + s) = pk.u;
    }
  }
}

// Flash attention. 4 waves/WG share one (b,h) and 64 queries; each 32-key
// K-tile (4KB) and V-tile (4KB) is staged once per WG in LDS (async-to-LDS on
// CDNA5 when available), then all score/PV fragments are ds_load_b128 reads.
__global__ void __launch_bounds__(128) attn_kernel(const __bf16* __restrict__ Q,
    const __bf16* __restrict__ K, const __bf16* __restrict__ VT,
    const float* __restrict__ padbias, float* __restrict__ pooled)
{
  const int tid = threadIdx.x;
  const int wave = tid >> 5;
  const int lane = tid & 31;
  const int g = lane >> 4, ln = lane & 15;
  const int qb = blockIdx.x & (Sv / QBLK - 1);  // 32 query blocks per (b,h)
  const int bh = blockIdx.x >> 5;
  const int h  = bh & (Hv - 1);
  const int b  = bh >> 3;
  const int q0 = qb * QBLK + wave * 16;         // this wave's 16 queries
  const int kend = (qb + 1) * QBLK;             // shared causal bound (multiple of 32)
  const float scale = 0.125f;                   // 1/sqrt(64)
  const float NEG = -__builtin_inff();

  __shared__ __align__(16) __bf16 ldsK[32 * 64];      // [key][dim]
  __shared__ __align__(16) __bf16 ldsV[64 * 32];      // [dim][key]
  __shared__ __align__(16) __bf16 ldsP[4][16 * 32];   // per-wave P staging

  const __bf16* qrow = Q + (b * Sv + q0 + ln) * Dv + h * HDv;
  v16bf aq0 = load_frag(qrow, g * 8,      16 + g * 8);
  v16bf aq1 = load_frag(qrow, 32 + g * 8, 48 + g * 8);

  float mrow[8], lrow[8], corr[8];
  v8f o0 = {}, o1 = {}, o2 = {}, o3 = {};
  #pragma unroll
  for (int r = 0; r < 8; ++r) { mrow[r] = NEG; lrow[r] = 0.f; }

  for (int k0 = 0; k0 < kend; k0 += 32) {
    // --- stage K-tile [32 keys x 64 dims] and V-tile [64 dims x 32 keys] in LDS
    #pragma unroll
    for (int j = 0; j < 2; ++j) {
      int c = tid + 128 * j;                   // 256 chunks of 16B = 4KB each tile
      int key = c >> 3, dc = c & 7;
      const __bf16* src = K + (b * Sv + k0 + key) * Dv + h * HDv + dc * 8;
#if HAVE_ASYNC
      __builtin_amdgcn_global_load_async_to_lds_b128(GPTR(src), LPTR(&ldsK[key * 64 + dc * 8]), 0, 0);
#else
      *(u32x4*)&ldsK[key * 64 + dc * 8] = *(const u32x4*)src;
#endif
      int dim = c >> 2, kc = c & 3;
      const __bf16* vsrc = VT + (b * Dv + h * HDv + dim) * Sv + k0 + kc * 8;
#if HAVE_ASYNC
      __builtin_amdgcn_global_load_async_to_lds_b128(GPTR(vsrc), LPTR(&ldsV[dim * 32 + kc * 8]), 0, 0);
#else
      *(u32x4*)&ldsV[dim * 32 + kc * 8] = *(const u32x4*)vsrc;
#endif
    }
    if (k0 + 32 < kend) {                      // prefetch next K-tile
      __builtin_prefetch(K + (b * Sv + k0 + 32 + (tid >> 2)) * Dv + h * HDv, 0, 1);
    }
#if HAVE_ASYNC
    __builtin_amdgcn_s_wait_asynccnt(0);
#endif
    __syncthreads();

    // --- scores: S = Q K^T, B-frags from ldsK
    v8f sf[2];
    float pbv[2];
    #pragma unroll
    for (int f = 0; f < 2; ++f) {
      const __bf16* krow = &ldsK[(16 * f + ln) * 64];
      v16bf kb0 = load_frag(krow, g * 8,      16 + g * 8);
      v16bf kb1 = load_frag(krow, 32 + g * 8, 48 + g * 8);
      v8f ss = {};
      ss = __builtin_amdgcn_wmma_f32_16x16x32_bf16(false, aq0, false, kb0, (short)0, ss, false, false);
      ss = __builtin_amdgcn_wmma_f32_16x16x32_bf16(false, aq1, false, kb1, (short)0, ss, false, false);
      sf[f] = ss;
      pbv[f] = padbias[b * Sv + k0 + 16 * f + ln];
    }
    // --- online softmax per query row (row spans a 16-lane group in C/D layout)
    #pragma unroll
    for (int r = 0; r < 8; ++r) {
      int qpos = q0 + r + 8 * g;
      float v0 = sf[0][r] * scale + pbv[0];
      float v1 = sf[1][r] * scale + pbv[1];
      if (k0 + ln > qpos)      v0 = NEG;       // causal
      if (k0 + 16 + ln > qpos) v1 = NEG;
      float mx = fmaxf(v0, v1);
      mx = fmaxf(mx, __shfl_xor(mx, 1, 16));
      mx = fmaxf(mx, __shfl_xor(mx, 2, 16));
      mx = fmaxf(mx, __shfl_xor(mx, 4, 16));
      mx = fmaxf(mx, __shfl_xor(mx, 8, 16));
      float mn = fmaxf(mrow[r], mx);
      float c  = (mrow[r] == NEG) ? 0.f : __expf(mrow[r] - mn);
      float p0 = (v0 == NEG) ? 0.f : __expf(v0 - mn);
      float p1 = (v1 == NEG) ? 0.f : __expf(v1 - mn);
      float ps = p0 + p1;
      ps += __shfl_xor(ps, 1, 16);
      ps += __shfl_xor(ps, 2, 16);
      ps += __shfl_xor(ps, 4, 16);
      ps += __shfl_xor(ps, 8, 16);
      lrow[r] = lrow[r] * c + ps;
      mrow[r] = mn;
      corr[r] = c;
      ldsP[wave][(r + 8 * g) * 32 + ln]      = (__bf16)p0;
      ldsP[wave][(r + 8 * g) * 32 + 16 + ln] = (__bf16)p1;
    }
    #pragma unroll
    for (int r = 0; r < 8; ++r) {
      o0[r] *= corr[r]; o1[r] *= corr[r]; o2[r] *= corr[r]; o3[r] *= corr[r];
    }
    // --- O += P @ V : P re-read in A layout (per-wave LDS), V B-frags from ldsV
    v16bf ap = load_frag(&ldsP[wave][ln * 32], g * 8, 16 + g * 8);
    {
      v16bf bv = load_frag(&ldsV[(0 * 16 + ln) * 32], g * 8, 16 + g * 8);
      o0 = __builtin_amdgcn_wmma_f32_16x16x32_bf16(false, ap, false, bv, (short)0, o0, false, false);
    }
    {
      v16bf bv = load_frag(&ldsV[(1 * 16 + ln) * 32], g * 8, 16 + g * 8);
      o1 = __builtin_amdgcn_wmma_f32_16x16x32_bf16(false, ap, false, bv, (short)0, o1, false, false);
    }
    {
      v16bf bv = load_frag(&ldsV[(2 * 16 + ln) * 32], g * 8, 16 + g * 8);
      o2 = __builtin_amdgcn_wmma_f32_16x16x32_bf16(false, ap, false, bv, (short)0, o2, false, false);
    }
    {
      v16bf bv = load_frag(&ldsV[(3 * 16 + ln) * 32], g * 8, 16 + g * 8);
      o3 = __builtin_amdgcn_wmma_f32_16x16x32_bf16(false, ap, false, bv, (short)0, o3, false, false);
    }
    __syncthreads();   // protect ldsK/ldsV before next iteration's staging
  }

  // Normalize per query and accumulate the S-mean into pooled[b, h*64 + n]
  const float invS = 1.0f / (float)Sv;
  #pragma unroll
  for (int t = 0; t < 4; ++t) {
    const v8f& o = (t == 0) ? o0 : (t == 1) ? o1 : (t == 2) ? o2 : o3;
    float acc = 0.f;
    #pragma unroll
    for (int r = 0; r < 8; ++r) {
      float l = lrow[r];
      acc += (l > 0.f) ? (o[r] / l) : 0.f;
    }
    atomicAdd(&pooled[b * Dv + h * HDv + t * 16 + ln], acc * invS);
  }
}

// Tiny head: po = pooled@Wo^T+bo ; hid = relu(po@W1^T+b1) ; out = hid@W2^T+b2
__global__ void __launch_bounds__(256) head_kernel(const float* __restrict__ pooled,
    const float* __restrict__ Wo, const float* __restrict__ bo,
    const float* __restrict__ W1, const float* __restrict__ b1,
    const float* __restrict__ W2, const float* __restrict__ b2,
    float* __restrict__ out)
{
  __shared__ float po[Bv * Dv];       // 16 KB
  __shared__ float hid[Bv * 2 * Dv];  // 32 KB
  const int tid = threadIdx.x;
  for (int i = tid; i < Bv * Dv; i += 256) {
    int b = i / Dv, n = i - b * Dv;
    float a = bo[n];
    const float* pr = pooled + b * Dv;
    const float* wr = Wo + n * Dv;
    for (int d = 0; d < Dv; ++d) a += pr[d] * wr[d];
    po[i] = a;
  }
  __syncthreads();
  for (int i = tid; i < Bv * 2 * Dv; i += 256) {
    int b = i / (2 * Dv), n = i - b * 2 * Dv;
    float a = b1[n];
    const float* pr = po + b * Dv;
    const float* wr = W1 + n * Dv;
    for (int d = 0; d < Dv; ++d) a += pr[d] * wr[d];
    hid[i] = fmaxf(a, 0.f);
  }
  __syncthreads();
  if (tid < Bv * C2) {
    int b = tid / C2, n = tid - b * C2;
    float a = b2[n];
    const float* hr = hid + b * 2 * Dv;
    const float* wr = W2 + n * 2 * Dv;
    for (int d = 0; d < 2 * Dv; ++d) a += hr[d] * wr[d];
    out[b * C2 + n] = a;
  }
}

extern "C" void kernel_launch(void* const* d_in, const int* in_sizes, int n_in,
                              void* d_out, int out_size, void* d_ws, size_t ws_size,
                              hipStream_t stream)
{
  const int*   x   = (const int*)  d_in[0];
  const float* emb = (const float*)d_in[1];
  const float* Wq  = (const float*)d_in[2];
  const float* bq  = (const float*)d_in[3];
  const float* Wk  = (const float*)d_in[4];
  const float* bk  = (const float*)d_in[5];
  const float* Wv  = (const float*)d_in[6];
  const float* bv  = (const float*)d_in[7];
  const float* Wo  = (const float*)d_in[8];
  const float* bo  = (const float*)d_in[9];
  const float* W1  = (const float*)d_in[10];
  const float* b1  = (const float*)d_in[11];
  const float* W2  = (const float*)d_in[12];
  const float* b2  = (const float*)d_in[13];
  const float* pe  = (const float*)d_in[14];

  char* ws = (char*)d_ws;
  size_t off = 0;
  auto alloc = [&](size_t bytes) -> char* {
    char* p = ws + off;
    off += (bytes + 255) & ~(size_t)255;
    return p;
  };
  __bf16* hbf  = (__bf16*)alloc((size_t)BSv * Dv * 2);   // 16 MB
  __bf16* Qb   = (__bf16*)alloc((size_t)BSv * Dv * 2);   // 16 MB
  __bf16* Kb   = (__bf16*)alloc((size_t)BSv * Dv * 2);   // 16 MB
  __bf16* VT   = (__bf16*)alloc((size_t)BSv * Dv * 2);   // 16 MB (transposed [B,D,S])
  __bf16* wqb  = (__bf16*)alloc((size_t)Dv * Dv * 2);
  __bf16* wkb  = (__bf16*)alloc((size_t)Dv * Dv * 2);
  __bf16* wvb  = (__bf16*)alloc((size_t)Dv * Dv * 2);
  float*  padb = (float*)alloc((size_t)BSv * 4);
  float*  pooled = (float*)alloc((size_t)Bv * Dv * 4);

  zero_kernel<<<(Bv * Dv + 255) / 256, 256, 0, stream>>>(pooled, Bv * Dv);
  cvt_kernel<<<(Dv * Dv + 255) / 256, 256, 0, stream>>>(Wq, wqb, Dv * Dv);
  cvt_kernel<<<(Dv * Dv + 255) / 256, 256, 0, stream>>>(Wk, wkb, Dv * Dv);
  cvt_kernel<<<(Dv * Dv + 255) / 256, 256, 0, stream>>>(Wv, wvb, Dv * Dv);
  embed_kernel<<<BSv, 128, 0, stream>>>(x, emb, pe, hbf, padb);
  qkv_kernel<<<(3 * (BSv / 16) * (Dv / 64)) / 4, 128, 0, stream>>>(
      hbf, wqb, wkb, wvb, bq, bk, bv, Qb, Kb, VT);
  attn_kernel<<<Bv * Hv * (Sv / QBLK), 128, 0, stream>>>(Qb, Kb, VT, padb, pooled);
  head_kernel<<<1, 256, 0, stream>>>(pooled, Wo, bo, W1, b1, W2, b2, (float*)d_out);
}
